// LSTMBlock_19224273617292
// MI455X (gfx1250) — matrix operand order repaired and checked by reading, not compiled, not run on hardware
//
#include <hip/hip_runtime.h>
#include <hip/hip_bf16.h>

// LSTM: B=512, T=512, D=128, H=64, gates 4H=256 (i,f,g,o)
// Phase P: prep  -> transpose W_ih/W_hh to [K][N] bf16, bias = b_ih + b_hh
// Phase A: xg = x @ W_ih^T + bias  (bf16 WMMA, f32 accum), stored as bf16
//          *in WMMA C-fragment order* tiled exactly as phase B consumes it.
// Phase B: 512-step recurrence; 4 waves share each 16-row group (H split 4-way),
//          W_hh^T in LDS, h re-swizzled through shared LDS staging each step.

typedef __attribute__((ext_vector_type(16))) __bf16 v16bf;
typedef __attribute__((ext_vector_type(8)))  __bf16 v8bf;
typedef __attribute__((ext_vector_type(8)))  float  v8f;

#define BATCH 512
#define TSTEPS 512
#define DIN 128
#define HID 64
#define NG 256  // 4*H
#define NBT 32  // BATCH/16 row groups

__device__ __forceinline__ __bf16 f2bf(float f) {
    union { float f; unsigned int u; } a; a.f = f;
    unsigned int r = a.u + 0x7FFFu + ((a.u >> 16) & 1u);  // round-to-nearest-even
    union { unsigned short s; __bf16 b; } o; o.s = (unsigned short)(r >> 16);
    return o.b;
}

#if __has_builtin(__builtin_amdgcn_rcpf)
#define FRCP(x) __builtin_amdgcn_rcpf(x)
#else
#define FRCP(x) (1.0f / (x))
#endif

__device__ __forceinline__ float fast_tanh(float x) {
#if __has_builtin(__builtin_amdgcn_tanhf)
    return __builtin_amdgcn_tanhf(x);            // v_tanh_f32 (CDNA5 TRANS op)
#elif __has_builtin(__builtin_amdgcn_tanh_f32)
    return __builtin_amdgcn_tanh_f32(x);
#else
    const float e = __expf(2.0f * x);            // v_exp_f32
    return 1.0f - 2.0f * FRCP(e + 1.0f);         // v_rcp_f32
#endif
}

__device__ __forceinline__ float fast_sigmoid(float x) {
    return 0.5f * fast_tanh(0.5f * x) + 0.5f;
}

// xg fragment address: tile (t, btile, ntile), per-lane 8 bf16 (16 B)
__device__ __forceinline__ size_t frag_idx(int t, int btile, int nt, int lane) {
    return ((((size_t)t * NBT + btile) * 16 + nt) * 32 + lane) * 8;
}

// ---------------- prep: weight transpose + bf16 convert + bias fold ----------------
__global__ void lstm_prep_kernel(const float* __restrict__ W_ih,
                                 const float* __restrict__ W_hh,
                                 const float* __restrict__ b_ih,
                                 const float* __restrict__ b_hh,
                                 __bf16* __restrict__ Wt_ih,   // [DIN][NG]
                                 __bf16* __restrict__ Wt_hh,   // [HID][NG]
                                 float* __restrict__ bias) {   // [NG]
    int idx = blockIdx.x * blockDim.x + threadIdx.x;
    int stride = gridDim.x * blockDim.x;
    for (int i = idx; i < DIN * NG; i += stride) {
        int k = i / NG, n = i % NG;
        Wt_ih[i] = f2bf(W_ih[(size_t)n * DIN + k]);
    }
    for (int i = idx; i < HID * NG; i += stride) {
        int k = i / NG, n = i % NG;
        Wt_hh[i] = f2bf(W_hh[(size_t)n * HID + k]);
    }
    if (idx < NG) bias[idx] = b_ih[idx] + b_hh[idx];
}

// ---------------- phase A: xg = x @ W_ih^T + bias ----------------
// One wave per (t, 16-batch-row) tile, all 16 N-tiles (A-frag reuse).
// Output stored bf16 in C-fragment order -> phase B reads 1x b128 per tile.
__global__ __launch_bounds__(256)
void lstm_xgemm_kernel(const float* __restrict__ x,       // [B][T][DIN]
                       const __bf16* __restrict__ Wt,     // [DIN][NG]
                       const float* __restrict__ bias,    // [NG]
                       __bf16* __restrict__ xgf) {        // fragment-ordered
    const int lane = threadIdx.x & 31;
    const int wave = threadIdx.x >> 5;
    const int tile = blockIdx.x * 8 + wave;   // 0 .. T*NBT-1
    const int t     = tile >> 5;              // timestep
    const int btile = tile & 31;              // 16-batch-row group
    const int mrow = lane & 15;               // A row / C-N col owned by this lane
    const int hi   = lane >> 4;

    // A fragments: 4 K-chunks of 32 (K=DIN=128), fp32 -> bf16 inline
    v16bf a[4];
    const float* xrow = x + ((size_t)(btile * 16 + mrow) * TSTEPS + t) * DIN;
    #pragma unroll
    for (int kc = 0; kc < 4; ++kc) {
        const int kb = kc * 32 + hi * 8;
        union { float4 q[4]; float s[16]; } u;
        u.q[0] = *(const float4*)(xrow + kb);
        u.q[1] = *(const float4*)(xrow + kb + 4);
        u.q[2] = *(const float4*)(xrow + kb + 16);
        u.q[3] = *(const float4*)(xrow + kb + 20);
        #pragma unroll
        for (int e = 0; e < 8; ++e) {
            a[kc][e]     = f2bf(u.s[e]);
            a[kc][e + 8] = f2bf(u.s[8 + e]);
        }
    }

    #pragma unroll 1
    for (int nt = 0; nt < 16; ++nt) {
        const int n0 = nt * 16;
        const float bval = bias[n0 + mrow];   // depends only on N
        v8f c;
        #pragma unroll
        for (int v = 0; v < 8; ++v) c[v] = bval;
        #pragma unroll
        for (int kc = 0; kc < 4; ++kc) {
            // B-frag: lane holds Wt[k][n0 + mrow], k in table order (kc*32 + hi*8 + s, +16)
            v16bf b;
            #pragma unroll
            for (int s = 0; s < 8; ++s) {
                b[s]     = Wt[(size_t)(kc * 32 + hi * 8 + s) * NG + n0 + mrow];
                b[s + 8] = Wt[(size_t)(kc * 32 + 16 + hi * 8 + s) * NG + n0 + mrow];
            }
            c = __builtin_amdgcn_wmma_f32_16x16x32_bf16(false, a[kc], false, b,
                                                        (short)0, c, false, false);
        }
        // one dense 16B store per lane (512B contiguous per wave)
        v8bf st;
        #pragma unroll
        for (int v = 0; v < 8; ++v) st[v] = f2bf(c[v]);
        *(v8bf*)(xgf + frag_idx(t, btile, nt, lane)) = st;
    }
}

// ---------------- phase B: recurrence ----------------
// 32 blocks x 4 waves. Block = one 16-batch-row group; each wave owns one
// 16-wide H-tile (j = wave): its i/f/g/o gate tiles + c/h slice. 128 waves total.
__global__ __launch_bounds__(128)
void lstm_rec_kernel(const __bf16* __restrict__ xgf,     // fragment-ordered
                     const __bf16* __restrict__ Wt_hh,   // [HID][NG]
                     float* __restrict__ out) {          // [B][HID]
    __shared__ __bf16 sWhh[HID * NG];      // 32 KB
    __shared__ __bf16 hs[16 * HID];        // 2 KB: h for this row group (shared by 4 waves)

    const int lane  = threadIdx.x & 31;
    const int j     = threadIdx.x >> 5;    // H-tile owned by this wave (0..3)
    const int btile = blockIdx.x;          // row group
    const int row0  = btile * 16;
    const int mrow  = lane & 15;
    const int hi    = lane >> 4;

    // Cooperative load of W_hh^T into LDS (32768 B = 2048 x uint4)
    {
        const uint4* src = (const uint4*)Wt_hh;
        uint4* dst = (uint4*)sWhh;
        for (int i = threadIdx.x; i < 2048; i += 128) dst[i] = src[i];
    }
    // h0 = 0
    {
        unsigned short* hz = (unsigned short*)hs;
        for (int i = threadIdx.x; i < 16 * HID; i += 128) hz[i] = 0;
    }
    __syncthreads();

    float cst[8], hn[8];
    #pragma unroll
    for (int v = 0; v < 8; ++v) { cst[v] = 0.0f; hn[v] = 0.0f; }

    #pragma unroll 1
    for (int t = 0; t < TSTEPS; ++t) {
        // h as A-fragments (full K=HID=64 -> 2 chunks) from shared staging
        v16bf a[2];
        const __bf16* hrow = hs + mrow * HID;
        #pragma unroll
        for (int kc = 0; kc < 2; ++kc) {
            const int kb = kc * 32 + hi * 8;
            union { v16bf v; v8bf h[2]; } u;
            u.h[0] = *(const v8bf*)(hrow + kb);
            u.h[1] = *(const v8bf*)(hrow + kb + 16);
            a[kc] = u.v;
        }

        v8f g4[4];
        #pragma unroll
        for (int p = 0; p < 4; ++p) {      // gates i,f,g,o at N-offsets p*64
            const int n0 = p * HID + j * 16;
            // xg tile as C operand: single coalesced b128 per lane
            const v8bf cf = *(const v8bf*)(xgf + frag_idx(t, btile, n0 >> 4, lane));
            v8f cc;
            #pragma unroll
            for (int v = 0; v < 8; ++v) cc[v] = (float)cf[v];
            #pragma unroll
            for (int kc = 0; kc < 2; ++kc) {
                v16bf b;
                #pragma unroll
                for (int s = 0; s < 8; ++s) {
                    b[s]     = sWhh[(size_t)(kc * 32 + hi * 8 + s) * NG + n0 + mrow];
                    b[s + 8] = sWhh[(size_t)(kc * 32 + 16 + hi * 8 + s) * NG + n0 + mrow];
                }
                cc = __builtin_amdgcn_wmma_f32_16x16x32_bf16(false, a[kc], false, b,
                                                             (short)0, cc, false, false);
            }
            g4[p] = cc;
            // prefetch next step's fragment for this gate tile
            if (t + 1 < TSTEPS)
                __builtin_prefetch(xgf + frag_idx(t + 1, btile, n0 >> 4, lane), 0, 0);
        }

        // column-local LSTM cell update (this wave's 16 H columns only)
        #pragma unroll
        for (int v = 0; v < 8; ++v) {
            const float iv = fast_sigmoid(g4[0][v]);
            const float fv = fast_sigmoid(g4[1][v]);
            const float gv = fast_tanh(g4[2][v]);
            const float ov = fast_sigmoid(g4[3][v]);
            const float cv = fv * cst[v] + iv * gv;
            cst[v] = cv;
            hn[v] = ov * fast_tanh(cv);
        }

        __syncthreads();   // all 4 waves finished reading hs for step t
        #pragma unroll
        for (int v = 0; v < 8; ++v) {
            const int m = v + hi * 8;
            hs[m * HID + j * 16 + mrow] = f2bf(hn[v]);   // publish this wave's H slice
        }
        __syncthreads();   // new h visible to all waves before step t+1
    }

    // Final hidden state -> out [B][H]
    #pragma unroll
    for (int v = 0; v < 8; ++v) {
        const int m = v + hi * 8;
        out[(size_t)(row0 + m) * HID + j * 16 + mrow] = hn[v];
    }
}

extern "C" void kernel_launch(void* const* d_in, const int* in_sizes, int n_in,
                              void* d_out, int out_size, void* d_ws, size_t ws_size,
                              hipStream_t stream) {
    (void)in_sizes; (void)n_in; (void)out_size; (void)ws_size;
    const float* x    = (const float*)d_in[0];   // [B,T,D]
    const float* W_ih = (const float*)d_in[1];   // [4H,D]
    const float* W_hh = (const float*)d_in[2];   // [4H,H]
    const float* b_ih = (const float*)d_in[3];   // [4H]
    const float* b_hh = (const float*)d_in[4];   // [4H]
    float* out = (float*)d_out;                  // [B,H]

    char* ws = (char*)d_ws;
    __bf16* Wt_ih = (__bf16*)(ws);               // 128*256 bf16 = 64 KB
    __bf16* Wt_hh = (__bf16*)(ws + 65536);       //  64*256 bf16 = 32 KB
    float*  bias  = (float*)(ws + 98304);        // 256 f32
    __bf16* xgf   = (__bf16*)(ws + 131072);      // 512*512*256 bf16 = 128 MB (frag order)

    lstm_prep_kernel<<<64, 256, 0, stream>>>(W_ih, W_hh, b_ih, b_hh, Wt_ih, Wt_hh, bias);
    // T*NBT = 16384 tiles of (t, 16 batch rows); 8 waves/block -> 2048 blocks
    lstm_xgemm_kernel<<<2048, 256, 0, stream>>>(x, Wt_ih, bias, xgf);
    // 32 row groups x 4 waves (H split 4-way) = 128 waves
    lstm_rec_kernel<<<32, 128, 0, stream>>>(xgf, Wt_hh, out);
}
